// ModulatedAttention_59047210385801
// MI455X (gfx1250) — matrix operand order repaired and checked by reading, not compiled, not run on hardware
//
#include <hip/hip_runtime.h>
#include <hip/hip_bf16.h>

typedef __bf16 bf16;
typedef __attribute__((ext_vector_type(16))) __bf16 v16bf;
typedef __attribute__((ext_vector_type(8)))  __bf16 v8bf;
typedef __attribute__((ext_vector_type(8)))  float  v8f;

#define NTOK 8192     // B*H*W
#define CIN  256
#define DHALF 128
#define HWN  1024     // H*W
#define NB   8
#define KDENSE 262144 // HW*C

// ---------- WMMA helpers (CDNA5 wave32, 16x16x32 bf16 -> f32) ----------

__device__ __forceinline__ v8f wmma_bf16(v16bf a, v16bf b, v8f c) {
  return __builtin_amdgcn_wmma_f32_16x16x32_bf16(false, a, false, b, (short)0, c,
                                                 false, false);
}

// A fragment (16x32): lane holds row (lane&15); elements i = K[(lane&16)+i],
// i.e. 16 contiguous bf16 along the contraction dim.  Two 16B chunks.
__device__ __forceinline__ v16bf load_a16(const bf16* p) {
  v8bf lo = *(const v8bf*)p;
  v8bf hi = *(const v8bf*)(p + 8);
  v16bf a;
#pragma unroll
  for (int i = 0; i < 8; ++i) { a[i] = lo[i]; a[i + 8] = hi[i]; }
  return a;
}

// B fragment (32x16): lane holds col (lane&15); elements i = K[(lane&16)+i],
// i.e. 16 contiguous bf16 (32B) along the contraction dim.
__device__ __forceinline__ v16bf load_b16(const bf16* p) {
  return *(const v16bf*)p;
}

// ---------- CDNA5 async Global->LDS copy (ASYNCcnt-tracked DMA path) ----------
// GV mode: per-lane 64-bit global address in VGPR pair, 32-bit LDS byte
// address in VDST VGPR (generic shared addr truncates to LDS offset per
// aperture rules).  16 bytes per lane per instruction.
__device__ __forceinline__ void async_copy16(void* lds_ptr, const void* gptr) {
  unsigned l = (unsigned)(unsigned long long)lds_ptr;
  asm volatile("global_load_async_to_lds_b128 %0, %1, off"
               :: "v"(l), "v"(gptr)
               : "memory");
}
__device__ __forceinline__ void wait_async0() {
  asm volatile("s_wait_asynccnt 0x0" ::: "memory");
}

// ---------- prep kernels ----------

__global__ __launch_bounds__(256) void k_cvt_bf16(const float* __restrict__ src,
                                                  bf16* __restrict__ dst, int n) {
  for (int i = blockIdx.x * blockDim.x + threadIdx.x; i < n;
       i += gridDim.x * blockDim.x)
    dst[i] = (bf16)src[i];
}

// dst[c*R + r] = src[r*C + c]  (f32 [R][C] -> bf16 [C][R])
__global__ __launch_bounds__(256) void k_tr_f32_bf16(const float* __restrict__ src,
                                                     bf16* __restrict__ dst,
                                                     int R, int C) {
  int n = R * C;
  for (int i = blockIdx.x * blockDim.x + threadIdx.x; i < n;
       i += gridDim.x * blockDim.x) {
    int r = i / C, c = i - r * C;
    dst[c * R + r] = (bf16)src[i];
  }
}

__global__ __launch_bounds__(256) void k_zero_f32(float* __restrict__ p, int n) {
  for (int i = blockIdx.x * blockDim.x + threadIdx.x; i < n;
       i += gridDim.x * blockDim.x)
    p[i] = 0.0f;
}

// ---------- projections: theta/phi/alpha = x @ W + b  (WMMA) ----------
// grid (512, 3); 8 waves/block, wave w owns N-tile w (128 cols / 8 tiles).
__global__ __launch_bounds__(256) void k_proj(
    const bf16* __restrict__ xb,
    const bf16* __restrict__ WtT, const bf16* __restrict__ WpT,
    const bf16* __restrict__ WaT,
    const float* __restrict__ bt, const float* __restrict__ bp,
    const float* __restrict__ ba,
    bf16* __restrict__ theta, bf16* __restrict__ phi,
    bf16* __restrict__ alphaT) {
  const int lane = threadIdx.x & 31, wave = threadIdx.x >> 5;
  const int l15 = lane & 15;
  const int koff = (lane & 16) >> 1;   // 0 or 8
  const int m0 = blockIdx.x * 16;
  const int n0 = wave * 16;
  const int sel = blockIdx.y;

  const bf16* WT = (sel == 0) ? WtT : (sel == 1) ? WpT : WaT;
  const float* bias = (sel == 0) ? bt : (sel == 1) ? bp : ba;

  const bf16* arow = xb + (size_t)(m0 + l15) * CIN;
  const bf16* brow = WT + (size_t)(n0 + l15) * CIN + (lane & 16);

  v8f acc = {};
#pragma unroll
  for (int ks = 0; ks < 8; ++ks) {   // K = 256 = 8 * 32
    v16bf A = load_a16(arow + ks * 32 + (lane & 16));
    v16bf B = load_b16(brow + ks * 32);
    acc = wmma_bf16(A, B, acc);
  }
  const float bv = bias[n0 + l15];

  if (sel < 2) {
    bf16* dst = (sel == 0) ? theta : phi;
#pragma unroll
    for (int r = 0; r < 8; ++r) {
      int row = m0 + r + koff;                     // koff doubles as +8 row shift
      dst[(size_t)row * DHALF + n0 + l15] = (bf16)(acc[r] + bv);
    }
  } else {
    // alphaT[col][token], col fixed per lane, tokens contiguous -> 16B store
    int col = n0 + l15;
    int tok0 = m0 + koff;
    v8bf pk;
#pragma unroll
    for (int r = 0; r < 8; ++r) pk[r] = (bf16)(acc[r] + bv);
    *(v8bf*)(alphaT + (size_t)col * NTOK + tok0) = pk;
  }
}

// ---------- flash attention: out = softmax(theta@phiT) @ alpha ----------
// grid 128 blocks x 4 waves = 512 query tiles of 16 rows.  The block streams
// 256 key blocks of 32 keys; phi/alphaT tiles are staged into LDS with
// double-buffered GLOBAL_LOAD_ASYNC_TO_LDS_B128 (shared by all 4 waves, copy
// of tile i+1 overlapped with the 16 WMMAs + online softmax of tile i).
__global__ __launch_bounds__(128) void k_flash(
    const bf16* __restrict__ theta, const bf16* __restrict__ phi,
    const bf16* __restrict__ alphaT, bf16* __restrict__ attnout) {
  __shared__ alignas(16) bf16 ph_s[2][32][128];   // [key][feat]  16 KB
  __shared__ alignas(16) bf16 aT_s[2][128][32];   // [feat][key]  16 KB
  __shared__ alignas(16) bf16 lds_p[4][16][32];   // P C->A layout swap, 4 KB

  const int tid = threadIdx.x;
  const int lane = tid & 31, wave = tid >> 5;
  const int l15 = lane & 15;
  const int hi = (lane & 16) >> 1;     // 0 or 8
  const int q0 = (blockIdx.x * 4 + wave) * 16;

  // stage key tile kb into buffer buf: phi tile is one contiguous 8KB block,
  // alphaT tile is 128 rows x 64B.  512+512 16B chunks over 128 threads.
  auto stage = [&](int buf, int kb) {
    const bf16* gph = phi + (size_t)kb * DHALF;
    bf16* lph = &ph_s[buf][0][0];
#pragma unroll
    for (int q = 0; q < 4; ++q) {
      int ci = q * 128 + tid;
      async_copy16(lph + ci * 8, gph + ci * 8);
    }
#pragma unroll
    for (int q = 0; q < 4; ++q) {
      int ci = q * 128 + tid;
      int row = ci >> 2, sub = ci & 3;
      async_copy16(&aT_s[buf][row][sub * 8],
                   alphaT + (size_t)row * NTOK + kb + sub * 8);
    }
  };

  // query tile A fragments, resident for the whole key loop (32 VGPRs)
  v16bf thA[4];
  const bf16* trow = theta + (size_t)(q0 + l15) * DHALF;
#pragma unroll
  for (int ks = 0; ks < 4; ++ks) thA[ks] = load_a16(trow + ks * 32 + (lane & 16));

  v8f acc[8];
  float rmax[8], rsum[8];
#pragma unroll
  for (int j = 0; j < 8; ++j) acc[j] = (v8f){};
#pragma unroll
  for (int r = 0; r < 8; ++r) { rmax[r] = -3.0e38f; rsum[r] = 0.0f; }

  stage(0, 0);                          // prologue copy of tile 0

  for (int kb = 0; kb < NTOK; kb += 32) {
    const int buf = (kb >> 5) & 1;
    wait_async0();                      // this wave's copies for tile kb done
    __syncthreads();                    // publish LDS; also WAR guard for buf
    if (kb + 32 < NTOK) stage(buf ^ 1, kb + 32);   // overlap next tile's DMA

    // S tile 16x32: two 16x16 C-frags, B operands from LDS
    v8f s0 = {}, s1 = {};
#pragma unroll
    for (int ks = 0; ks < 4; ++ks) {
      s0 = wmma_bf16(thA[ks],
                     load_b16(&ph_s[buf][l15][ks * 32 + (lane & 16)]), s0);
      s1 = wmma_bf16(thA[ks],
                     load_b16(&ph_s[buf][16 + l15][ks * 32 + (lane & 16)]), s1);
    }

    // online softmax per row (row r+hi lives across the 16-lane half-group)
#pragma unroll
    for (int r = 0; r < 8; ++r) {
      float mx = fmaxf(s0[r], s1[r]);
      mx = fmaxf(mx, __shfl_xor(mx, 1, 32));
      mx = fmaxf(mx, __shfl_xor(mx, 2, 32));
      mx = fmaxf(mx, __shfl_xor(mx, 4, 32));
      mx = fmaxf(mx, __shfl_xor(mx, 8, 32));
      float nm = fmaxf(rmax[r], mx);
      float corr = __expf(rmax[r] - nm);
      float p0 = __expf(s0[r] - nm);
      float p1 = __expf(s1[r] - nm);
      float ls = p0 + p1;
      ls += __shfl_xor(ls, 1, 32);
      ls += __shfl_xor(ls, 2, 32);
      ls += __shfl_xor(ls, 4, 32);
      ls += __shfl_xor(ls, 8, 32);
      rsum[r] = rsum[r] * corr + ls;
      rmax[r] = nm;
#pragma unroll
      for (int j = 0; j < 8; ++j) acc[j][r] *= corr;
      lds_p[wave][r + hi][l15] = (bf16)p0;
      lds_p[wave][r + hi][l15 + 16] = (bf16)p1;
    }
    __syncthreads();                    // P tile visible across the wave group

    v16bf P = load_a16(&lds_p[wave][l15][lane & 16]);
#pragma unroll
    for (int j = 0; j < 8; ++j)
      acc[j] = wmma_bf16(P, load_b16(&aT_s[buf][j * 16 + l15][lane & 16]),
                         acc[j]);
    // next iteration's top barrier is the WAR guard for lds_p and buf
  }

#pragma unroll
  for (int j = 0; j < 8; ++j)
#pragma unroll
    for (int r = 0; r < 8; ++r)
      attnout[(size_t)(q0 + r + hi) * DHALF + j * 16 + l15] =
          (bf16)(acc[j][r] / rsum[r]);
}

// ---------- dense spatial-gate GEMV: logits[b][n] += x[b,:] . Wd[:,n] ----------
// grid (8 n-tiles of 128, 64 K-chunks of 4096). 1.07 GB of W_dense streamed
// once -> HBM-bandwidth bound, VALU FMA with LDS-staged x.
__global__ __launch_bounds__(256) void k_dense(const float* __restrict__ x,
                                               const float* __restrict__ Wd,
                                               float* __restrict__ logits) {
  __shared__ float xs[8][1024];
  const int tid = threadIdx.x;
  const int col = blockIdx.x * 128 + (tid & 127);
  const int b0 = (tid >> 7) * 4;
  const size_t kbase = (size_t)blockIdx.y * 4096;

  float a0 = 0.f, a1 = 0.f, a2 = 0.f, a3 = 0.f;
  for (int sc = 0; sc < 4; ++sc) {
    size_t k0 = kbase + (size_t)sc * 1024;
    __syncthreads();
    for (int i = tid; i < 8192; i += 256) {
      int b = i >> 10, kk = i & 1023;
      xs[b][kk] = x[(size_t)b * KDENSE + k0 + kk];
    }
    __syncthreads();
    for (int kk = 0; kk < 1024; ++kk) {
      float w = Wd[(k0 + kk) * 1024 + col];
      a0 = fmaf(xs[b0 + 0][kk], w, a0);
      a1 = fmaf(xs[b0 + 1][kk], w, a1);
      a2 = fmaf(xs[b0 + 2][kk], w, a2);
      a3 = fmaf(xs[b0 + 3][kk], w, a3);
    }
  }
  atomicAdd(&logits[(b0 + 0) * HWN + col], a0);
  atomicAdd(&logits[(b0 + 1) * HWN + col], a1);
  atomicAdd(&logits[(b0 + 2) * HWN + col], a2);
  atomicAdd(&logits[(b0 + 3) * HWN + col], a3);
}

// ---------- softmax over HW positions per batch ----------
__global__ __launch_bounds__(256) void k_spsm(const float* __restrict__ logits,
                                              const float* __restrict__ bd,
                                              float* __restrict__ sp) {
  __shared__ float red[8];
  const int b = blockIdx.x, tid = threadIdx.x;
  const int lane = tid & 31, wave = tid >> 5;

  float v[4];
#pragma unroll
  for (int i = 0; i < 4; ++i) {
    int c = tid + i * 256;
    v[i] = logits[b * HWN + c] + bd[c];
  }
  float mx = fmaxf(fmaxf(v[0], v[1]), fmaxf(v[2], v[3]));
  for (int off = 16; off >= 1; off >>= 1) mx = fmaxf(mx, __shfl_xor(mx, off, 32));
  if (lane == 0) red[wave] = mx;
  __syncthreads();
  float bmax = red[0];
#pragma unroll
  for (int w = 1; w < 8; ++w) bmax = fmaxf(bmax, red[w]);
  __syncthreads();

  float s = 0.f;
#pragma unroll
  for (int i = 0; i < 4; ++i) { v[i] = __expf(v[i] - bmax); s += v[i]; }
  for (int off = 16; off >= 1; off >>= 1) s += __shfl_xor(s, off, 32);
  if (lane == 0) red[wave] = s;
  __syncthreads();
  float bsum = 0.f;
#pragma unroll
  for (int w = 0; w < 8; ++w) bsum += red[w];
  float inv = 1.0f / bsum;
#pragma unroll
  for (int i = 0; i < 4; ++i) sp[b * HWN + tid + i * 256] = v[i] * inv;
}

// ---------- final: out = (attnout @ W_mask) * sp + x  (WMMA) ----------
// grid (512, 2); 8 waves/block; wave owns one 16-col tile of the 256 cols.
__global__ __launch_bounds__(256) void k_final(
    const bf16* __restrict__ ao, const bf16* __restrict__ WmT,
    const float* __restrict__ sp, const float* __restrict__ x,
    float* __restrict__ out) {
  const int lane = threadIdx.x & 31, wave = threadIdx.x >> 5;
  const int l15 = lane & 15;
  const int koff = (lane & 16) >> 1;
  const int m0 = blockIdx.x * 16;
  const int n0 = (blockIdx.y * 8 + wave) * 16;

  const bf16* arow = ao + (size_t)(m0 + l15) * DHALF;
  const bf16* brow = WmT + (size_t)(n0 + l15) * DHALF + (lane & 16);

  v8f acc = {};
#pragma unroll
  for (int ks = 0; ks < 4; ++ks) {   // K = 128
    v16bf A = load_a16(arow + ks * 32 + (lane & 16));
    v16bf B = load_b16(brow + ks * 32);
    acc = wmma_bf16(A, B, acc);
  }

  const int col = n0 + l15;
#pragma unroll
  for (int r = 0; r < 8; ++r) {
    int m = m0 + r + koff;
    float g = sp[(m >> 10) * HWN + (m & 1023)];
    size_t idx = (size_t)m * CIN + col;
    out[idx] = acc[r] * g + x[idx];
  }
}

// ---------- launch ----------

extern "C" void kernel_launch(void* const* d_in, const int* in_sizes, int n_in,
                              void* d_out, int out_size, void* d_ws, size_t ws_size,
                              hipStream_t stream) {
  const float* x  = (const float*)d_in[0];
  const float* Wt = (const float*)d_in[1];
  const float* bt = (const float*)d_in[2];
  const float* Wp = (const float*)d_in[3];
  const float* bp = (const float*)d_in[4];
  const float* Wa = (const float*)d_in[5];
  const float* ba = (const float*)d_in[6];
  const float* Wm = (const float*)d_in[7];
  const float* Wd = (const float*)d_in[8];
  const float* bd = (const float*)d_in[9];
  float* out = (float*)d_out;

  char* ws = (char*)d_ws;
  size_t off = 0;
  auto carve = [&](size_t bytes) -> char* {
    char* p = ws + off;
    off += (bytes + 255) & ~(size_t)255;
    return p;
  };
  bf16* xb     = (bf16*)carve((size_t)NTOK * CIN * 2);    // x in bf16
  bf16* th     = (bf16*)carve((size_t)NTOK * DHALF * 2);  // theta
  bf16* ph     = (bf16*)carve((size_t)NTOK * DHALF * 2);  // phi
  bf16* aT     = (bf16*)carve((size_t)DHALF * NTOK * 2);  // alpha transposed
  bf16* ao     = (bf16*)carve((size_t)NTOK * DHALF * 2);  // attention out
  bf16* WtT    = (bf16*)carve((size_t)DHALF * CIN * 2);
  bf16* WpT    = (bf16*)carve((size_t)DHALF * CIN * 2);
  bf16* WaT    = (bf16*)carve((size_t)DHALF * CIN * 2);
  bf16* WmT    = (bf16*)carve((size_t)CIN * DHALF * 2);
  float* logit = (float*)carve((size_t)NB * HWN * 4);
  float* sp    = (float*)carve((size_t)NB * HWN * 4);
  (void)in_sizes; (void)n_in; (void)out_size; (void)ws_size;

  k_cvt_bf16<<<1024, 256, 0, stream>>>(x, xb, NTOK * CIN);
  k_tr_f32_bf16<<<64, 256, 0, stream>>>(Wt, WtT, CIN, DHALF);
  k_tr_f32_bf16<<<64, 256, 0, stream>>>(Wp, WpT, CIN, DHALF);
  k_tr_f32_bf16<<<64, 256, 0, stream>>>(Wa, WaT, CIN, DHALF);
  k_tr_f32_bf16<<<64, 256, 0, stream>>>(Wm, WmT, DHALF, CIN);
  k_zero_f32<<<32, 256, 0, stream>>>(logit, NB * HWN);

  k_proj<<<dim3(NTOK / 16, 3), 256, 0, stream>>>(xb, WtT, WpT, WaT, bt, bp, ba,
                                                 th, ph, aT);
  k_flash<<<128, 128, 0, stream>>>(th, ph, aT, ao);
  k_dense<<<dim3(8, 64), 256, 0, stream>>>(x, Wd, logit);
  k_spsm<<<NB, 256, 0, stream>>>(logit, bd, sp);
  k_final<<<dim3(NTOK / 16, 2), 256, 0, stream>>>(ao, WmT, sp, x, out);
}
